// GCNLayer_25537875542345
// MI455X (gfx1250) — matrix operand order repaired and checked by reading, not compiled, not run on hardware
//
#include <hip/hip_runtime.h>
#include <hip/hip_bf16.h>

typedef __attribute__((ext_vector_type(16))) __bf16 v16bf;
typedef __attribute__((ext_vector_type(8)))  float  v8f;

#define N_NODES 325
#define A_COLS  352          // padded K extent of a_norm (11 chunks of 32)
#define A_ROWS  336          // padded M extent (21 m-tiles)
#define XT_PITCH 360         // LDS pitch for x^T rows (bank-conflict-free, 16B aligned)
#define WT_PITCH 72          // LDS pitch for W^T rows
#define Z_PITCH  68          // LDS pitch for z scratch rows (floats)
#define MTILES   21
#define BN_COUNT 665600.0f   // B*N*T = 32*325*64

union frag_u { v16bf v; uint4 q[2]; };

// A-fragment: lanes 0-15 K={0..7,16..23}, lanes 16-31 K={8..15,24..31}
__device__ __forceinline__ v16bf ldA(const __bf16* p) {
    frag_u u;
    u.q[0] = *(const uint4*)p;        // 8 bf16
    u.q[1] = *(const uint4*)(p + 16); // 8 bf16, +16 elements
    return u.v;
}
// B-fragment: lane = column, lanes 0-15 K={0..7,16..23}, lanes 16-31 K={8..15,24..31}
__device__ __forceinline__ v16bf ldB(const __bf16* p) {
    frag_u u;
    u.q[0] = *(const uint4*)p;
    u.q[1] = *(const uint4*)(p + 16);
    return u.v;
}
// convert 8+8 fp32 (from LDS z scratch) into an A-fragment
__device__ __forceinline__ v16bf cvtA(const float* p0, const float* p1) {
    float4 a = *(const float4*)p0;
    float4 b = *(const float4*)(p0 + 4);
    float4 c = *(const float4*)p1;
    float4 d = *(const float4*)(p1 + 4);
    v16bf v;
    v[0]=(__bf16)a.x; v[1]=(__bf16)a.y; v[2]=(__bf16)a.z; v[3]=(__bf16)a.w;
    v[4]=(__bf16)b.x; v[5]=(__bf16)b.y; v[6]=(__bf16)b.z; v[7]=(__bf16)b.w;
    v[8]=(__bf16)c.x; v[9]=(__bf16)c.y; v[10]=(__bf16)c.z; v[11]=(__bf16)c.w;
    v[12]=(__bf16)d.x; v[13]=(__bf16)d.y; v[14]=(__bf16)d.z; v[15]=(__bf16)d.w;
    return v;
}

// ---- Kernel 1: dis[m] = rsqrt(rowsum(A)+1) ----------------------------------
__global__ void gcn_deg(const float* __restrict__ adj, float* __restrict__ dis) {
    __shared__ float red[128];
    const int m = blockIdx.x;
    float s = 0.f;
    for (int n = threadIdx.x; n < N_NODES; n += 128) s += adj[m * N_NODES + n];
    red[threadIdx.x] = s;
    __syncthreads();
    for (int off = 64; off > 0; off >>= 1) {
        if (threadIdx.x < off) red[threadIdx.x] += red[threadIdx.x + off];
        __syncthreads();
    }
    if (threadIdx.x == 0) {
        float deg = red[0] + 1.0f;
        dis[m] = deg > 0.f ? rsqrtf(deg) : 0.f;
    }
}

// ---- Kernel 2: a_norm (bf16, zero-padded to 336x352) + zero BN stats --------
__global__ void gcn_anorm(const float* __restrict__ adj, const float* __restrict__ dis,
                          __bf16* __restrict__ anorm, float* __restrict__ stats) {
    const int id = blockIdx.x * 256 + threadIdx.x;
    if (id < 128) stats[id] = 0.f;
    if (id >= A_ROWS * A_COLS) return;
    const int m = id / A_COLS, n = id - m * A_COLS;
    float v = 0.f;
    if (m < N_NODES && n < N_NODES) {
        float a = adj[m * N_NODES + n] + (m == n ? 1.f : 0.f);
        v = dis[m] * a * dis[n];
    }
    anorm[id] = (__bf16)v;
}

// ---- Kernel 3: fused  out[b,:,t,:] = (a_norm @ x[b,:,t,:]) @ W  + BN sums ---
__global__ __launch_bounds__(128) void gcn_main(
        const float* __restrict__ x, const float* __restrict__ w,
        const __bf16* __restrict__ anorm, float* __restrict__ out,
        float* __restrict__ stats) {
    __shared__ __bf16 xT[64 * XT_PITCH];   // x^T slice: [f][n], bf16
    __shared__ __bf16 WT[64 * WT_PITCH];   // W^T:       [o][f], bf16
    __shared__ float  zS[4 * 16 * Z_PITCH];// per-wave z scratch: [m][f]
    __shared__ float  st[128];             // BN sums / sumsq

    const int tid = threadIdx.x;
    const int b = blockIdx.x >> 6;
    const int t = blockIdx.x & 63;

    st[tid] = 0.f;

    // stage W^T into LDS (coalesced over o)
    for (int i = tid; i < 64 * 64; i += 128) {
        int f = i >> 6, o = i & 63;
        WT[o * WT_PITCH + f] = (__bf16)w[f * 64 + o];
    }
    // stage x[b,:,t,:]^T into LDS, zero-pad n to 360 (coalesced over f)
    const float* xb = x + (size_t)b * N_NODES * 4096 + (size_t)t * 64;
    for (int i = tid; i < XT_PITCH * 64; i += 128) {
        int n = i >> 6, f = i & 63;
        float v = (n < N_NODES) ? xb[(size_t)n * 4096 + f] : 0.f;
        xT[f * XT_PITCH + n] = (__bf16)v;
    }
    __syncthreads();

    const int wave = tid >> 5, lane = tid & 31;
    const int lhi = lane >> 4, llo = lane & 15;
    float* zw = zS + wave * 16 * Z_PITCH;

    float bns[4]  = {0.f, 0.f, 0.f, 0.f};
    float bns2[4] = {0.f, 0.f, 0.f, 0.f};

    for (int mt = wave; mt < MTILES; mt += 4) {
        // ---- stage 1: z(16x64) = a_norm[mt] (16x352) @ x^T chunks -----------
        v8f acc[4] = {};
        const __bf16* arow = anorm + (size_t)(mt * 16 + llo) * A_COLS;
        for (int kc = 0; kc < 11; ++kc) {
            const int ka = kc * 32 + lhi * 8;   // A K-base for this lane half
            const int kb = kc * 32 + lhi * 8;   // B K-base for this lane half
            v16bf afrag = ldA(arow + ka);
#pragma unroll
            for (int ft = 0; ft < 4; ++ft) {
                v16bf bfrag = ldB(&xT[(ft * 16 + llo) * XT_PITCH + kb]);
                acc[ft] = __builtin_amdgcn_wmma_f32_16x16x32_bf16(
                    false, afrag, false, bfrag, (short)0, acc[ft], false, false);
            }
        }
        // spill z (C-layout) to per-wave LDS so it can be re-read in A-layout
#pragma unroll
        for (int ft = 0; ft < 4; ++ft)
#pragma unroll
            for (int j = 0; j < 8; ++j)
                zw[(j + lhi * 8) * Z_PITCH + ft * 16 + llo] = acc[ft][j];

        // ---- stage 2: out(16x64) = z(16x64) @ W(64x64) ----------------------
        v8f acc2[4] = {};
#pragma unroll
        for (int kc = 0; kc < 2; ++kc) {
            const float* zp = zw + llo * Z_PITCH + kc * 32 + lhi * 8;
            v16bf a2 = cvtA(zp, zp + 16);
            const int kb2 = kc * 32 + lhi * 8;
#pragma unroll
            for (int ot = 0; ot < 4; ++ot) {
                v16bf b2 = ldB(&WT[(ot * 16 + llo) * WT_PITCH + kb2]);
                acc2[ot] = __builtin_amdgcn_wmma_f32_16x16x32_bf16(
                    false, a2, false, b2, (short)0, acc2[ot], false, false);
            }
        }
        // ---- epilogue: store pre-BN xc, accumulate channel sums -------------
#pragma unroll
        for (int ot = 0; ot < 4; ++ot) {
#pragma unroll
            for (int j = 0; j < 8; ++j) {
                int m = mt * 16 + j + lhi * 8;
                if (m < N_NODES) {
                    float v = acc2[ot][j];
                    out[(((size_t)b * N_NODES + m) * 64 + t) * 64 + ot * 16 + llo] = v;
                    bns[ot]  += v;
                    bns2[ot] += v * v;
                }
            }
        }
    }
    // BN reduction: registers -> LDS -> global
#pragma unroll
    for (int ot = 0; ot < 4; ++ot) {
        atomicAdd(&st[ot * 16 + llo], bns[ot]);
        atomicAdd(&st[64 + ot * 16 + llo], bns2[ot]);
    }
    __syncthreads();
    atomicAdd(&stats[tid], st[tid]);
}

// ---- Kernel 4: per-channel scale/shift from batch stats ---------------------
__global__ void gcn_bnparams(const float* __restrict__ stats,
                             const float* __restrict__ gamma,
                             const float* __restrict__ beta,
                             float* __restrict__ scsh) {
    const int c = threadIdx.x;
    const float inv_cnt = 1.0f / BN_COUNT;
    float mean = stats[c] * inv_cnt;
    float var  = stats[64 + c] * inv_cnt - mean * mean;
    float inv  = rsqrtf(var + 1e-5f);
    float sc   = gamma[c] * inv;
    scsh[c]      = sc;
    scsh[64 + c] = beta[c] - mean * sc;
}

// ---- Kernel 5: in-place BN apply + ReLU (float4) ----------------------------
__global__ void gcn_bnrelu(float* __restrict__ out, const float* __restrict__ scsh, int n4) {
    const int i = blockIdx.x * 256 + threadIdx.x;
    if (i >= n4) return;
    float4 v = ((const float4*)out)[i];
    const int c0 = (i * 4) & 63;  // rows are 64-channel aligned
    v.x = fmaxf(fmaf(v.x, scsh[c0 + 0], scsh[64 + c0 + 0]), 0.f);
    v.y = fmaxf(fmaf(v.y, scsh[c0 + 1], scsh[64 + c0 + 1]), 0.f);
    v.z = fmaxf(fmaf(v.z, scsh[c0 + 2], scsh[64 + c0 + 2]), 0.f);
    v.w = fmaxf(fmaf(v.w, scsh[c0 + 3], scsh[64 + c0 + 3]), 0.f);
    ((float4*)out)[i] = v;
}

extern "C" void kernel_launch(void* const* d_in, const int* in_sizes, int n_in,
                              void* d_out, int out_size, void* d_ws, size_t ws_size,
                              hipStream_t stream) {
    const float* x     = (const float*)d_in[0];  // (32,325,64,64)
    const float* adj   = (const float*)d_in[1];  // (325,325)
    const float* wgt   = (const float*)d_in[2];  // (64,64)
    const float* gamma = (const float*)d_in[3];  // (64,)
    const float* beta  = (const float*)d_in[4];  // (64,)
    float* out = (float*)d_out;

    // workspace layout
    char* ws = (char*)d_ws;
    float*  stats = (float*)(ws + 0);      // 128 f32: sums | sumsq
    float*  scsh  = (float*)(ws + 512);    // 128 f32: scale | shift
    float*  dis   = (float*)(ws + 1024);   // 325 f32
    __bf16* anorm = (__bf16*)(ws + 2560);  // 336*352 bf16 (~231 KB)

    gcn_deg<<<N_NODES, 128, 0, stream>>>(adj, dis);
    gcn_anorm<<<(A_ROWS * A_COLS + 255) / 256, 256, 0, stream>>>(adj, dis, anorm, stats);
    gcn_main<<<32 * 64, 128, 0, stream>>>(x, wgt, anorm, out, stats);
    gcn_bnparams<<<1, 64, 0, stream>>>(stats, gamma, beta, scsh);
    const int n4 = out_size / 4;
    gcn_bnrelu<<<(n4 + 255) / 256, 256, 0, stream>>>(out, scsh, n4);
}
